// GRURegressor_74062416052719
// MI455X (gfx1250) — matrix operand order repaired and checked by reading, not compiled, not run on hardware
//
#include <hip/hip_runtime.h>

// ---------------------------------------------------------------------------
// CDNA5 (gfx1250) GRU regressor: bf16 WMMA (f32 accum) for all GEMMs.
//   B=64, T=512, I=2048, H=128, 3H=384.
// ---------------------------------------------------------------------------

typedef __attribute__((ext_vector_type(16))) __bf16     v16bf;
typedef __attribute__((ext_vector_type(8)))  float      v8f;
typedef __attribute__((ext_vector_type(4)))  unsigned   u4;
typedef __attribute__((ext_vector_type(4)))  __bf16     bf4;

union FragCast { u4 u[2]; v16bf v; };

// A-matrix fragment (16x32 bf16) from row-major bf16 in LDS.
// Layout: lane (l=lane&15,hi=lane>>4): VGPR0-3 = A[l, kb+hi*16 .. +7],
// VGPR4-7 = A[l, kb+hi*16+8 .. +7]  -> two contiguous 16-byte chunks.
__device__ __forceinline__ v16bf afrag_lds_bf16(const __bf16* base, int ld,
                                                int kb, int lane) {
  const int l = lane & 15, hi = lane >> 4;
  const __bf16* p = base + l * ld + kb + hi * 16;
  FragCast f;
  f.u[0] = *(const u4*)(p);
  f.u[1] = *(const u4*)(p + 8);
  return f.v;
}

// B-matrix fragment (32x16 bf16) from row-major bf16 in LDS, where the
// stored matrix is W [N x K] and B = W^T.  lane l -> N=l; element i -> K=kb+hi*16+i.
__device__ __forceinline__ v16bf bfrag_lds_bf16(const __bf16* base, int ld,
                                                int kb, int lane) {
  const int l = lane & 15, hi = lane >> 4;
  const __bf16* p = base + l * ld + kb + hi * 16;
  FragCast f;
  f.u[0] = *(const u4*)(p);
  f.u[1] = *(const u4*)(p + 8);
  return f.v;
}

// B fragment from global f32 row-major W [N x K] (B = W^T), cvt to bf16.
__device__ __forceinline__ v16bf bfrag_gbl_f32(const float* __restrict__ W,
                                               long ld, int nbase, int kb, int lane) {
  const int l = lane & 15, hi = lane >> 4;
  const float* p = W + (long)(nbase + l) * ld + kb + hi * 16;
  v16bf r;
#pragma unroll
  for (int i = 0; i < 16; ++i) r[i] = (__bf16)p[i];
  return r;
}

__device__ __forceinline__ float sigf(float v) {
  return 1.f / (1.f + __expf(-v));
}

// ---------------------------------------------------------------------------
// Kernel 1: xg[B*T,384] = x[B*T,2048] @ w_ih0^T + b_ih0     (51.5 GFLOP)
//
// LDS-tiled: block = 8 waves (256 thr), output 64(M) x 128(N), K-chunks of 256.
// Each chunk: cooperative f32->bf16 staging of A(64x256) and W(128x256) into
// LDS (each element converted ONCE per block), then 8 k-steps x (1 B-frag +
// 4 A-frags + 4 WMMA) per wave. Rows padded to 264 (132 DWORDs == 4 banks)
// so fragment loads are bank-conflict-free.
// Grid: N-block fastest (3 per M group) -> x rows stay L2-hot across the
// 3 blocks that share them; HBM traffic ~= x read once (268 MB, ~12 us floor).
// ---------------------------------------------------------------------------
#define K1_LDA 264                      // padded row stride (bf16 elements)
#define K1_SMEM ((64 + 128) * K1_LDA * 2)  // 101376 bytes

__global__ __launch_bounds__(256) void k_xproj(const float* __restrict__ x,
                                               const float* __restrict__ w_ih0,
                                               const float* __restrict__ b_ih0,
                                               float* __restrict__ xg) {
  extern __shared__ unsigned char smem1[];
  __bf16* atile = (__bf16*)smem1;          // [64][K1_LDA]
  __bf16* wtile = atile + 64 * K1_LDA;     // [128][K1_LDA]

  const int tid  = threadIdx.x;
  const int lane = tid & 31;
  const int wv   = tid >> 5;               // 0..7 -> N-tile within block
  const int l    = lane & 15, hi = lane >> 4;
  const int nb   = blockIdx.x % 3;         // N block (128 cols of 384)
  const long mbase = (long)(blockIdx.x / 3) * 64;

  v8f z = {};
  v8f acc[4] = {z, z, z, z};

  for (int kc = 0; kc < 2048; kc += 256) {
    // ---- stage A chunk: 64 rows x 256 cols (4096 float4 units) ----
    for (int u = tid; u < 64 * 64; u += 256) {
      const int row = u >> 6, c4 = (u & 63) << 2;
      float4 f = *(const float4*)(x + (mbase + row) * 2048 + kc + c4);
      bf4 v = {(__bf16)f.x, (__bf16)f.y, (__bf16)f.z, (__bf16)f.w};
      *(bf4*)(atile + row * K1_LDA + c4) = v;
    }
    // ---- stage W chunk: 128 rows x 256 cols (8192 float4 units) ----
    for (int u = tid; u < 128 * 64; u += 256) {
      const int row = u >> 6, c4 = (u & 63) << 2;
      float4 f = *(const float4*)(w_ih0 + (long)(nb * 128 + row) * 2048 + kc + c4);
      bf4 v = {(__bf16)f.x, (__bf16)f.y, (__bf16)f.z, (__bf16)f.w};
      *(bf4*)(wtile + row * K1_LDA + c4) = v;
    }
    __syncthreads();

    // ---- compute: 8 k-steps x 4 M-tiles per wave ----
#pragma unroll
    for (int ks = 0; ks < 8; ++ks) {
      v16bf b = bfrag_lds_bf16(wtile + wv * 16 * K1_LDA, K1_LDA, ks * 32, lane);
#pragma unroll
      for (int mt = 0; mt < 4; ++mt) {
        v16bf a = afrag_lds_bf16(atile + mt * 16 * K1_LDA, K1_LDA, ks * 32, lane);
        acc[mt] = __builtin_amdgcn_wmma_f32_16x16x32_bf16(false, a, false, b,
                                                          (short)0, acc[mt],
                                                          false, false);
      }
    }
    __syncthreads();
  }

  // D layout: element r -> row hi*8+r, col l.
  const int col = nb * 128 + wv * 16 + l;
  const float bias = b_ih0[col];
#pragma unroll
  for (int mt = 0; mt < 4; ++mt) {
    float* dst = xg + (mbase + mt * 16 + hi * 8) * 384 + col;
#pragma unroll
    for (int r = 0; r < 8; ++r) dst[(long)r * 384] = acc[mt][r] + bias;
  }
}

// ---------------------------------------------------------------------------
// Kernel 2: fused 2-layer GRU scan + FC head.
// 4 workgroups, each owns 16 independent batch rows. 24 waves; wave w owns
// gate columns [16w,16w+16). Weight WMMA B-fragments stay resident in VGPRs
// for all 512 time steps; only the bf16 hidden state flows through LDS.
// ---------------------------------------------------------------------------
__global__ __launch_bounds__(768) void k_gru_scan(
    const float* __restrict__ xg,
    const float* __restrict__ w_hh0, const float* __restrict__ b_hh0,
    const float* __restrict__ w_ih1, const float* __restrict__ b_ih1,
    const float* __restrict__ w_hh1, const float* __restrict__ b_hh1,
    const float* __restrict__ w_fc1, const float* __restrict__ b_fc1,
    const float* __restrict__ w_fc2, const float* __restrict__ b_fc2,
    float* __restrict__ out) {
  extern __shared__ unsigned char smem[];
  float*  gx  = (float*)smem;                  // [16][384] x-side gates (L1)
  float*  gh  = gx + 16 * 384;                 // [16][384] h-side gates (L0/L1)
  float*  h0f = gh + 16 * 384;                 // [16][128] layer-0 h (f32)
  float*  h2f = h0f + 16 * 128;                // [16][128] layer-1 h (f32)
  __bf16* h0b = (__bf16*)(h2f + 16 * 128);     // [16][128] layer-0 h (bf16)
  __bf16* h2b = h0b + 16 * 128;                // [16][128] layer-1 h (bf16)

  const int tid   = threadIdx.x;
  const int lane  = tid & 31;
  const int wv    = tid >> 5;                  // 0..23 -> N-tile
  const int l     = lane & 15, hi = lane >> 4;
  const int bbase = blockIdx.x * 16;           // batch rows of this WG

  // Resident weight fragments for the three recurrent GEMMs (K=128 -> 4 steps).
  v16bf Bh0[4], Bx1[4], Bh1[4];
#pragma unroll
  for (int k = 0; k < 4; ++k) {
    Bh0[k] = bfrag_gbl_f32(w_hh0, 128, wv * 16, k * 32, lane);
    Bx1[k] = bfrag_gbl_f32(w_ih1, 128, wv * 16, k * 32, lane);
    Bh1[k] = bfrag_gbl_f32(w_hh1, 128, wv * 16, k * 32, lane);
  }

  for (int e = tid; e < 16 * 128; e += (int)blockDim.x) {
    h0f[e] = 0.f; h2f[e] = 0.f;
    h0b[e] = (__bf16)0.f; h2b[e] = (__bf16)0.f;
  }
  __syncthreads();

  for (int t = 0; t < 512; ++t) {
    // -------- Phase A: gh = h0 @ w_hh0^T  (16x384x128) --------
    {
      v8f acc = {};
#pragma unroll
      for (int k = 0; k < 4; ++k) {
        v16bf a = afrag_lds_bf16(h0b, 128, k * 32, lane);
        acc = __builtin_amdgcn_wmma_f32_16x16x32_bf16(false, a, false, Bh0[k],
                                                      (short)0, acc, false, false);
      }
#pragma unroll
      for (int r = 0; r < 8; ++r)
        gh[(hi * 8 + r) * 384 + wv * 16 + l] = acc[r];
    }
    __syncthreads();

    // -------- Activation layer 0: h0 <- GRU(xg[t], gh, h0) --------
    for (int e = tid; e < 16 * 128; e += (int)blockDim.x) {
      const int m = e >> 7, c = e & 127;
      const float* xrow = xg + ((long)(bbase + m) * 512 + t) * 384;
      const float hr = gh[m * 384 + c]       + b_hh0[c];
      const float hz = gh[m * 384 + 128 + c] + b_hh0[128 + c];
      const float hn = gh[m * 384 + 256 + c] + b_hh0[256 + c];
      const float r  = sigf(xrow[c] + hr);
      const float z  = sigf(xrow[128 + c] + hz);
      const float n  = tanhf(xrow[256 + c] + r * hn);
      const float hv = (1.f - z) * n + z * h0f[e];
      h0f[e] = hv;
      h0b[e] = (__bf16)hv;                     // layer-1 input == new h0
    }
    __syncthreads();

    // -------- Phase B: gx = h1 @ w_ih1^T ; gh = h2 @ w_hh1^T --------
    {
      v8f ax = {}, ah = {};
#pragma unroll
      for (int k = 0; k < 4; ++k) {
        v16bf a0 = afrag_lds_bf16(h0b, 128, k * 32, lane);
        ax = __builtin_amdgcn_wmma_f32_16x16x32_bf16(false, a0, false, Bx1[k],
                                                     (short)0, ax, false, false);
        v16bf a2 = afrag_lds_bf16(h2b, 128, k * 32, lane);
        ah = __builtin_amdgcn_wmma_f32_16x16x32_bf16(false, a2, false, Bh1[k],
                                                     (short)0, ah, false, false);
      }
#pragma unroll
      for (int r = 0; r < 8; ++r) {
        gx[(hi * 8 + r) * 384 + wv * 16 + l] = ax[r];
        gh[(hi * 8 + r) * 384 + wv * 16 + l] = ah[r];
      }
    }
    __syncthreads();

    // -------- Activation layer 1: h2 <- GRU(gx, gh, h2) --------
    for (int e = tid; e < 16 * 128; e += (int)blockDim.x) {
      const int m = e >> 7, c = e & 127;
      const float xr = gx[m * 384 + c]       + b_ih1[c];
      const float xz = gx[m * 384 + 128 + c] + b_ih1[128 + c];
      const float xn = gx[m * 384 + 256 + c] + b_ih1[256 + c];
      const float hr = gh[m * 384 + c]       + b_hh1[c];
      const float hz = gh[m * 384 + 128 + c] + b_hh1[128 + c];
      const float hn = gh[m * 384 + 256 + c] + b_hh1[256 + c];
      const float r  = sigf(xr + hr);
      const float z  = sigf(xz + hz);
      const float n  = tanhf(xn + r * hn);
      const float hv = (1.f - z) * n + z * h2f[e];
      h2f[e] = hv;
      h2b[e] = (__bf16)hv;
    }
    __syncthreads();
  }

  // -------- FC head: relu(h2 @ w_fc1^T + b_fc1) @ w_fc2^T + b_fc2 --------
  for (int e = tid; e < 16 * 64; e += (int)blockDim.x) {
    const int m = e >> 6, j = e & 63;
    float s = b_fc1[j];
    const float* hrow = h2f + m * 128;
    const float* wrow = w_fc1 + j * 128;
#pragma unroll 8
    for (int k2 = 0; k2 < 128; ++k2) s += hrow[k2] * wrow[k2];
    gx[e] = s > 0.f ? s : 0.f;
  }
  __syncthreads();
  if (tid < 16) {
    float s = b_fc2[0];
    const float* yrow = gx + tid * 64;
#pragma unroll
    for (int j = 0; j < 64; ++j) s += yrow[j] * w_fc2[j];
    out[bbase + tid] = s;
  }
}

// ---------------------------------------------------------------------------
extern "C" void kernel_launch(void* const* d_in, const int* in_sizes, int n_in,
                              void* d_out, int out_size, void* d_ws, size_t ws_size,
                              hipStream_t stream) {
  (void)in_sizes; (void)n_in; (void)out_size; (void)ws_size;
  const float* x     = (const float*)d_in[0];
  const float* w_ih0 = (const float*)d_in[1];
  const float* w_hh0 = (const float*)d_in[2];
  const float* b_ih0 = (const float*)d_in[3];
  const float* b_hh0 = (const float*)d_in[4];
  const float* w_ih1 = (const float*)d_in[5];
  const float* w_hh1 = (const float*)d_in[6];
  const float* b_ih1 = (const float*)d_in[7];
  const float* b_hh1 = (const float*)d_in[8];
  const float* w_fc1 = (const float*)d_in[9];
  const float* b_fc1 = (const float*)d_in[10];
  const float* w_fc2 = (const float*)d_in[11];
  const float* b_fc2 = (const float*)d_in[12];

  float* xg = (float*)d_ws;  // [64*512, 384] f32 = 48 MB

  // Input projection: 512 M-blocks x 3 N-blocks (N fastest for L2 reuse).
  (void)hipFuncSetAttribute((const void*)k_xproj,
                            hipFuncAttributeMaxDynamicSharedMemorySize, K1_SMEM);
  k_xproj<<<dim3(3 * 512), dim3(256), K1_SMEM, stream>>>(x, w_ih0, b_ih0, xg);

  const int smem_bytes = (2 * 16 * 384 + 2 * 16 * 128) * 4 + (2 * 16 * 128) * 2; // 73728
  (void)hipFuncSetAttribute((const void*)k_gru_scan,
                            hipFuncAttributeMaxDynamicSharedMemorySize, smem_bytes);
  k_gru_scan<<<dim3(4), dim3(768), smem_bytes, stream>>>(
      xg, w_hh0, b_hh0, w_ih1, b_ih1, w_hh1, b_hh1,
      w_fc1, b_fc1, w_fc2, b_fc2, (float*)d_out);
}
